// GraphNN_59459527246043
// MI455X (gfx1250) — matrix operand order repaired and checked by reading, not compiled, not run on hardware
//
#include <hip/hip_runtime.h>
#include <math.h>

// ---------------------------------------------------------------------------
// Problem constants (from reference)
// ---------------------------------------------------------------------------
#define BS    512
#define NNODE 64                    // N = MJ + MM
#define D_    128
#define H_    5
#define NTOT  32768                 // BS * NNODE
#define E_CNT 92160                 // BS * MJ * K
#define HD_   640                   // H_ * D_

typedef __attribute__((ext_vector_type(16))) __bf16 v16bf;
typedef __attribute__((ext_vector_type(8)))  float  v8f;

// fp32 x2 -> packed bf16x2 in one dword via the native CDNA5 pack instruction
// (clang emits this mnemonic itself for scalar converts on gfx1250, but has no
// declared builtin on this toolchain, so use inline asm for the 2-wide form)
__device__ __forceinline__ unsigned int pack2bf(float x, float y) {
    unsigned int r;
    asm("v_cvt_pk_bf16_f32 %0, %1, %2" : "=v"(r) : "v"(x), "v"(y));
    return r;
}

__device__ __forceinline__ void atomicMaxF(float* addr, float val) {
    if (val >= 0.0f) atomicMax((int*)addr, __float_as_int(val));
    else             atomicMin((unsigned int*)addr, __float_as_uint(val));
}

// ---------------------------------------------------------------------------
// Generic GEMM:  C(M,N) = act( A(M,K) @ W(N,K)^T + bias )
// bf16 WMMA (v_wmma_f32_16x16x32_bf16), fp32 accumulate.
// Block: 256 threads (8 waves). Block tile 128x128, wave tile 32x64,
// K-step 32, double-buffered LDS, global_prefetch for K+2 tiles.
// ACT: 0 = none, 2 = relu
// Requires: M % 128 == 0, N % 128 == 0, K % 32 == 0.
// ---------------------------------------------------------------------------
union FragU { uint4 q[2]; v16bf v; };

template <int ACT>
__global__ __launch_bounds__(256) void gemm_bias_act(
    const float* __restrict__ A, const float* __restrict__ W,
    const float* __restrict__ bias, float* __restrict__ C,
    int M, int N, int K)
{
    // 2 buffers x (128x32) bf16 for A and W tiles -> 32 KB total
    __shared__ __align__(16) unsigned short AsBuf[2 * 128 * 32];
    __shared__ __align__(16) unsigned short BsBuf[2 * 128 * 32];

    const int m0   = blockIdx.x * 128;
    const int n0   = blockIdx.y * 128;
    const int t    = threadIdx.x;
    const int wave = t >> 5;
    const int lane = t & 31;
    const int half = lane >> 4;      // 0 or 1
    const int l16  = lane & 15;
    const int wr   = wave & 3;       // wave row: C rows [32*wr, 32*wr+32)
    const int wc   = wave >> 2;      // wave col: C cols [64*wc, 64*wc+64)

    v8f acc[2][4];
#pragma unroll
    for (int i = 0; i < 2; ++i)
#pragma unroll
        for (int j = 0; j < 4; ++j)
            acc[i][j] = (v8f){0.f,0.f,0.f,0.f,0.f,0.f,0.f,0.f};

    float4 av[4], bv[4];
    auto loadTiles = [&](int k0) {
#pragma unroll
        for (int i = 0; i < 4; ++i) {
            int f = t + i * 256;        // float4 index within 128x8 tile
            int r = f >> 3, c4 = f & 7; // row, float4-column
            av[i] = *(const float4*)&A[(size_t)(m0 + r) * K + k0 + c4 * 4];
            bv[i] = *(const float4*)&W[(size_t)(n0 + r) * K + k0 + c4 * 4];
        }
    };
    auto storeTiles = [&](int buf) {
        unsigned short* As = AsBuf + buf * (128 * 32);
        unsigned short* Bs = BsBuf + buf * (128 * 32);
#pragma unroll
        for (int i = 0; i < 4; ++i) {
            int f = t + i * 256;
            int r = f >> 3, c4 = f & 7;
            uint2 pa, pb;
            pa.x = pack2bf(av[i].x, av[i].y); pa.y = pack2bf(av[i].z, av[i].w);
            pb.x = pack2bf(bv[i].x, bv[i].y); pb.y = pack2bf(bv[i].z, bv[i].w);
            *(uint2*)&As[r * 32 + c4 * 4] = pa;
            *(uint2*)&Bs[r * 32 + c4 * 4] = pb;
        }
    };

    const int nsteps = K >> 5;
    loadTiles(0);
    for (int s = 0; s < nsteps; ++s) {
        const int buf = s & 1;
        storeTiles(buf);
        __syncthreads();
        if (s + 1 < nsteps) loadTiles((s + 1) << 5);
        if (s + 2 < nsteps) {
            // pull the K+2 tiles toward the WGP (global_prefetch_b8)
            int k2 = (s + 2) << 5;
            int pr = t & 127;
            __builtin_prefetch(&A[(size_t)(m0 + pr) * K + k2], 0, 1);
            __builtin_prefetch(&W[(size_t)(n0 + pr) * K + k2], 0, 1);
        }

        const unsigned short* As = AsBuf + buf * (128 * 32);
        const unsigned short* Bs = BsBuf + buf * (128 * 32);

        // A fragments (16x32 bf16, ISA layout): lane<16 -> K 0..7 / 16..23
        const int am0 = (2 * wr + 0) * 16 + l16;
        const int am1 = (2 * wr + 1) * 16 + l16;
        FragU a0, a1;
        a0.q[0] = *(const uint4*)&As[am0 * 32 + half * 8];
        a0.q[1] = *(const uint4*)&As[am0 * 32 + 16 + half * 8];
        a1.q[0] = *(const uint4*)&As[am1 * 32 + half * 8];
        a1.q[1] = *(const uint4*)&As[am1 * 32 + 16 + half * 8];

#pragma unroll
        for (int j = 0; j < 4; ++j) {
            // B fragment (32x16 = W^T tile): lane<16 -> K 0..7 / 16..23
            const int bn = (4 * wc + j) * 16 + l16;
            FragU bf_;
            bf_.q[0] = *(const uint4*)&Bs[bn * 32 + half * 8];
            bf_.q[1] = *(const uint4*)&Bs[bn * 32 + 16 + half * 8];
            acc[0][j] = __builtin_amdgcn_wmma_f32_16x16x32_bf16(
                false, a0.v, false, bf_.v, (short)0, acc[0][j], false, false);
            acc[1][j] = __builtin_amdgcn_wmma_f32_16x16x32_bf16(
                false, a1.v, false, bf_.v, (short)0, acc[1][j], false, false);
        }
    }

    // store: C/D layout: VGPR r -> row r + 8*half, col = l16
#pragma unroll
    for (int mi = 0; mi < 2; ++mi) {
#pragma unroll
        for (int j = 0; j < 4; ++j) {
            const int n = n0 + (4 * wc + j) * 16 + l16;
            const float bvv = bias ? bias[n] : 0.f;
#pragma unroll
            for (int i = 0; i < 8; ++i) {
                int m = m0 + (2 * wr + mi) * 16 + half * 8 + i;
                float v = acc[mi][j][i] + bvv;
                if (ACT == 2) v = fmaxf(v, 0.f);
                C[(size_t)m * N + n] = v;
            }
        }
    }
}

// ---------------------------------------------------------------------------
// EdgeGAT helpers
// ---------------------------------------------------------------------------
// el/er per node per head: one wave per head, grid = NTOT, block = 160
__global__ __launch_bounds__(160) void node_attn_scores(
    const float* __restrict__ feat, const float* __restrict__ attn_l,
    const float* __restrict__ attn_r, float* __restrict__ el, float* __restrict__ er)
{
    const int n = blockIdx.x;
    const int h = threadIdx.x >> 5;
    const int lane = threadIdx.x & 31;
    float al = 0.f, ar = 0.f;
    for (int d = lane; d < D_; d += 32) {
        float f = feat[(size_t)n * HD_ + h * D_ + d];
        al += f * attn_l[h * D_ + d];
        ar += f * attn_r[h * D_ + d];
    }
    for (int o = 16; o > 0; o >>= 1) { al += __shfl_down(al, o); ar += __shfl_down(ar, o); }
    if (lane == 0) { el[n * H_ + h] = al; er[n * H_ + h] = ar; }
}

// a_e[h] = sum_d attn_e[h,d] * W_e[h*D + d]   (single block of 160 threads)
__global__ __launch_bounds__(160) void compute_ae(
    const float* __restrict__ attn_e, const float* __restrict__ W_e, float* __restrict__ ae)
{
    const int h = threadIdx.x >> 5, lane = threadIdx.x & 31;
    float s = 0.f;
    for (int d = lane; d < D_; d += 32) s += attn_e[h * D_ + d] * W_e[h * D_ + d];
    for (int o = 16; o > 0; o >>= 1) s += __shfl_down(s, o);
    if (lane == 0) ae[h] = s;
}

__global__ void fillk(float* __restrict__ p, float v, size_t n) {
    size_t i = (size_t)blockIdx.x * 256 + threadIdx.x;
    if (i < n) p[i] = v;
}

// pass 1: e = leaky_relu(el[src]+er[dst]+ef*ae, 0.2); segment-max over dst
__global__ void edge_score(
    const float* __restrict__ el, const float* __restrict__ er,
    const float* __restrict__ ef, const float* __restrict__ ae,
    const int* __restrict__ src, const int* __restrict__ dst,
    float* __restrict__ ee, float* __restrict__ mm)
{
    int idx = blockIdx.x * 256 + threadIdx.x;
    if (idx >= E_CNT * H_) return;
    int i = idx / H_, h = idx - i * H_;
    float e = el[src[i] * H_ + h] + er[dst[i] * H_ + h] + ef[i] * ae[h];
    e = (e > 0.f) ? e : 0.2f * e;
    ee[idx] = e;
    atomicMaxF(&mm[dst[i] * H_ + h], e);
}

// pass 2: ex = exp(e - m[dst]); segment-sum over dst
__global__ void edge_exp(
    const float* __restrict__ ee, const float* __restrict__ mm,
    const int* __restrict__ dst, float* __restrict__ ex, float* __restrict__ ss)
{
    int idx = blockIdx.x * 256 + threadIdx.x;
    if (idx >= E_CNT * H_) return;
    int i = idx / H_, h = idx - i * H_;
    float e = __expf(ee[idx] - mm[dst[i] * H_ + h]);
    ex[idx] = e;
    atomicAdd(&ss[dst[i] * H_ + h], e);
}

// pass 3: out[dst,h,:] += feat[src,h,:] * (ex/s[dst]); one lane handles 4 dims
__global__ void edge_scatter(
    const float* __restrict__ feat, const float* __restrict__ ex,
    const float* __restrict__ ss, const int* __restrict__ src,
    const int* __restrict__ dst, float* __restrict__ outb)
{
    size_t idx = (size_t)blockIdx.x * 256 + threadIdx.x;
    if (idx >= (size_t)E_CNT * H_ * 32) return;
    int lane = (int)(idx & 31);
    size_t eh = idx >> 5;
    int i = (int)(eh / H_), h = (int)(eh - (size_t)i * H_);
    float a = ex[eh] / ss[dst[i] * H_ + h];
    size_t db = (size_t)dst[i] * HD_ + h * D_;
    size_t sb = (size_t)src[i] * HD_ + h * D_;
#pragma unroll
    for (int dd = 0; dd < 4; ++dd) {
        int d = lane + dd * 32;
        atomicAdd(&outb[db + d], feat[sb + d] * a);
    }
}

// bias + leaky_relu(0.01) in place, and mean over heads -> xmean (next conv input)
__global__ __launch_bounds__(128) void bias_act_mean(
    float* __restrict__ outb, const float* __restrict__ bias, float* __restrict__ xmean)
{
    size_t n = blockIdx.x; int d = threadIdx.x;
    float s = 0.f;
#pragma unroll
    for (int h = 0; h < H_; ++h) {
        size_t idx = n * HD_ + h * D_ + d;
        float v = outb[idx] + bias[h * D_ + d];
        v = (v > 0.f) ? v : 0.01f * v;
        outb[idx] = v;
        s += v;
    }
    xmean[n * D_ + d] = s * 0.2f;
}

// ---------------------------------------------------------------------------
// Decoder: fused attention for one (batch, head); Sq = Sk = 64, hd = 16.
// ---------------------------------------------------------------------------
__global__ __launch_bounds__(256) void mha_attn(
    const float* __restrict__ q, int ldq, int qoff,
    const float* __restrict__ k, int ldk, int koff,
    const float* __restrict__ v, int ldv, int voff,
    float* __restrict__ o)
{
    const int bh = blockIdx.x;
    const int b = bh >> 3, h = bh & 7;
    __shared__ float qs[64][16], ks[64][16], vs[64][16];
    __shared__ float sc[64][64];
    const int t = threadIdx.x;
    const size_t rowb = (size_t)b * 64;

    for (int i = t; i < 1024; i += 256) {
        int r = i >> 4, c = i & 15;
        qs[r][c] = q[(rowb + r) * ldq + qoff + h * 16 + c];
        ks[r][c] = k[(rowb + r) * ldk + koff + h * 16 + c];
        vs[r][c] = v[(rowb + r) * ldv + voff + h * 16 + c];
    }
    __syncthreads();

    for (int i = t; i < 4096; i += 256) {
        int r = i >> 6, c = i & 63;
        float s = 0.f;
#pragma unroll
        for (int d = 0; d < 16; ++d) s += qs[r][d] * ks[c][d];
        sc[r][c] = s * 0.25f;     // 1/sqrt(16)
    }
    __syncthreads();

    if (t < 64) {
        float mx = -1e30f;
        for (int c = 0; c < 64; ++c) mx = fmaxf(mx, sc[t][c]);
        float sm = 0.f;
        for (int c = 0; c < 64; ++c) { float e = __expf(sc[t][c] - mx); sc[t][c] = e; sm += e; }
        float inv = 1.f / sm;
        for (int c = 0; c < 64; ++c) sc[t][c] *= inv;
    }
    __syncthreads();

    for (int i = t; i < 1024; i += 256) {
        int r = i >> 4, d = i & 15;
        float s = 0.f;
        for (int c = 0; c < 64; ++c) s += sc[r][c] * vs[c][d];
        o[(rowb + r) * D_ + h * 16 + d] = s;
    }
}

// out = LayerNorm(out + add) * g + b    (one row of 128 per block)
__global__ __launch_bounds__(128) void ln_res(
    float* __restrict__ out, const float* __restrict__ add,
    const float* __restrict__ g, const float* __restrict__ b)
{
    const int r = blockIdx.x, t = threadIdx.x;
    __shared__ float red[128];
    size_t base = (size_t)r * D_;
    float v = out[base + t] + add[base + t];
    red[t] = v; __syncthreads();
    for (int o = 64; o > 0; o >>= 1) { if (t < o) red[t] += red[t + o]; __syncthreads(); }
    float mu = red[0] * (1.f / 128.f);
    __syncthreads();
    float dv = v - mu;
    red[t] = dv * dv; __syncthreads();
    for (int o = 64; o > 0; o >>= 1) { if (t < o) red[t] += red[t + o]; __syncthreads(); }
    float var = red[0] * (1.f / 128.f);
    out[base + t] = dv * rsqrtf(var + 1e-5f) * g[t] + b[t];
}

// ---------------------------------------------------------------------------
// Host driver
// Assumed d_in layout: top-level dict insertion order, nested params flattened
// jax-style (dict keys sorted):
//   0 encoder_output, 1 edge_feats, 2 src, 3 dst,
//   convs[c] (c=0..2) at 4+6c: [W, W_e, attn_e, attn_l, attn_r, b]
//   dec[l]   (l=0..2) at 22+18l: [W1, W2, Wo_c, Wo_s, Wqkv_c, Wqkv_s,
//                                 b1, b2, bo_c, bo_s, bqkv_c, bqkv_s,
//                                 ln1g, ln1b, ln2g, ln2b, ln3g, ln3b]
//   76 lin_W, 77 lin_b
// ---------------------------------------------------------------------------
extern "C" void kernel_launch(void* const* d_in, const int* in_sizes, int n_in,
                              void* d_out, int out_size, void* d_ws, size_t ws_size,
                              hipStream_t stream)
{
    (void)in_sizes; (void)n_in; (void)out_size; (void)ws_size;

    const float* enc = (const float*)d_in[0];
    const float* ef  = (const float*)d_in[1];
    const int*   src = (const int*)d_in[2];
    const int*   dst = (const int*)d_in[3];
    auto F = [&](int i) { return (const float*)d_in[i]; };

    float* ws  = (float*)d_ws;
    float* out = (float*)d_out;

    // workspace arena (floats); region A (conv feat) is reused as qkv/o/tmp,
    // region B (conv accum) is reused as ff/kv.
    const size_t FEAT_SZ = (size_t)NTOT * HD_;               // 20,971,520
    const size_t REGB_SZ = (size_t)NTOT * 512 + (size_t)NTOT * 256;
    float* feat  = ws;
    float* outb  = ws + FEAT_SZ;
    float* qkv   = ws;
    float* obuf  = ws + (size_t)NTOT * 384;
    float* tmpb  = ws + (size_t)NTOT * 512;                  // 384+128
    float* ffb   = outb;
    float* kvb   = outb + (size_t)NTOT * 512;
    float* xmean = ws + FEAT_SZ + REGB_SZ;
    float* elb   = xmean + (size_t)NTOT * D_;
    float* erb   = elb + (size_t)NTOT * H_;
    float* mmb   = erb + (size_t)NTOT * H_;
    float* ssb   = mmb + (size_t)NTOT * H_;
    float* aeb   = ssb + (size_t)NTOT * H_;
    float* eeb   = aeb + 8;
    float* exb   = eeb + (size_t)E_CNT * H_;

    const size_t NH = (size_t)NTOT * H_;       // 163,840
    const size_t EH = (size_t)E_CNT * H_;      // 460,800

    // ---------------- EdgeGAT convs ----------------
    const float* xin = enc;
    for (int c = 0; c < 3; ++c) {
        int pb = 4 + 6 * c;
        const float* W   = F(pb + 0);
        const float* W_e = F(pb + 1);
        const float* a_e = F(pb + 2);
        const float* a_l = F(pb + 3);
        const float* a_r = F(pb + 4);
        const float* bb  = F(pb + 5);

        gemm_bias_act<0><<<dim3(NTOT / 128, HD_ / 128), 256, 0, stream>>>(
            xin, W, nullptr, feat, NTOT, HD_, D_);
        node_attn_scores<<<NTOT, 160, 0, stream>>>(feat, a_l, a_r, elb, erb);
        compute_ae<<<1, 160, 0, stream>>>(a_e, W_e, aeb);
        fillk<<<(unsigned)((FEAT_SZ + 255) / 256), 256, 0, stream>>>(outb, 0.f, FEAT_SZ);
        fillk<<<(unsigned)((NH + 255) / 256), 256, 0, stream>>>(mmb, -1e30f, NH);
        fillk<<<(unsigned)((NH + 255) / 256), 256, 0, stream>>>(ssb, 0.f, NH);
        edge_score<<<(unsigned)((EH + 255) / 256), 256, 0, stream>>>(
            elb, erb, ef, aeb, src, dst, eeb, mmb);
        edge_exp<<<(unsigned)((EH + 255) / 256), 256, 0, stream>>>(eeb, mmb, dst, exb, ssb);
        edge_scatter<<<(unsigned)((EH * 32 + 255) / 256), 256, 0, stream>>>(
            feat, exb, ssb, src, dst, outb);
        bias_act_mean<<<NTOT, 128, 0, stream>>>(outb, bb, xmean);
        xin = xmean;
    }

    // ---------------- linear projection -> decoder state (in d_out) ----------
    gemm_bias_act<0><<<dim3(NTOT / 128, D_ / 128), 256, 0, stream>>>(
        outb, F(76), F(77), out, NTOT, D_, HD_);

    // ---------------- decoder layers ----------------
    for (int l = 0; l < 3; ++l) {
        int pb = 22 + 18 * l;
        const float* W1     = F(pb + 0);
        const float* W2     = F(pb + 1);
        const float* Wo_c   = F(pb + 2);
        const float* Wo_s   = F(pb + 3);
        const float* Wqkv_c = F(pb + 4);
        const float* Wqkv_s = F(pb + 5);
        const float* b1     = F(pb + 6);
        const float* b2     = F(pb + 7);
        const float* bo_c   = F(pb + 8);
        const float* bo_s   = F(pb + 9);
        const float* bq_c   = F(pb + 10);
        const float* bq_s   = F(pb + 11);
        const float* ln1g   = F(pb + 12);
        const float* ln1b   = F(pb + 13);
        const float* ln2g   = F(pb + 14);
        const float* ln2b   = F(pb + 15);
        const float* ln3g   = F(pb + 16);
        const float* ln3b   = F(pb + 17);

        // self attention
        gemm_bias_act<0><<<dim3(NTOT / 128, 384 / 128), 256, 0, stream>>>(
            out, Wqkv_s, bq_s, qkv, NTOT, 384, D_);
        mha_attn<<<BS * 8, 256, 0, stream>>>(qkv, 384, 0, qkv, 384, 128, qkv, 384, 256, obuf);
        gemm_bias_act<0><<<dim3(NTOT / 128, D_ / 128), 256, 0, stream>>>(
            obuf, Wo_s, bo_s, tmpb, NTOT, D_, D_);
        ln_res<<<NTOT, 128, 0, stream>>>(out, tmpb, ln1g, ln1b);

        // cross attention (q from out, k/v from encoder_output)
        gemm_bias_act<0><<<dim3(NTOT / 128, D_ / 128), 256, 0, stream>>>(
            out, Wqkv_c, bq_c, qkv, NTOT, D_, D_);
        gemm_bias_act<0><<<dim3(NTOT / 128, 256 / 128), 256, 0, stream>>>(
            enc, Wqkv_c + 128 * 128, bq_c + 128, kvb, NTOT, 256, D_);
        mha_attn<<<BS * 8, 256, 0, stream>>>(qkv, 128, 0, kvb, 256, 0, kvb, 256, 128, obuf);
        gemm_bias_act<0><<<dim3(NTOT / 128, D_ / 128), 256, 0, stream>>>(
            obuf, Wo_c, bo_c, tmpb, NTOT, D_, D_);
        ln_res<<<NTOT, 128, 0, stream>>>(out, tmpb, ln2g, ln2b);

        // FFN
        gemm_bias_act<2><<<dim3(NTOT / 128, 512 / 128), 256, 0, stream>>>(
            out, W1, b1, ffb, NTOT, 512, D_);
        gemm_bias_act<0><<<dim3(NTOT / 128, D_ / 128), 256, 0, stream>>>(
            ffb, W2, b2, tmpb, NTOT, D_, 512);
        ln_res<<<NTOT, 128, 0, stream>>>(out, tmpb, ln3g, ln3b);
    }
}
